// _Block_240518168940
// MI455X (gfx1250) — matrix operand order repaired and checked by reading, not compiled, not run on hardware
//
#include <hip/hip_runtime.h>

#define DIM 768
#define HEADS 12
#define HEAD_DIM 64
#define HIDDEN 3072
#define SEQ 2048
#define BATCH 4
#define ROWS (BATCH * SEQ)   // 8192

typedef __attribute__((ext_vector_type(16))) __bf16 v16bf;
typedef __attribute__((ext_vector_type(8)))  float  v8f;

union BF16Frag {
    v16bf v;
    uint4 u[2];
    unsigned short s[16];
};

__device__ __forceinline__ unsigned short f2bf(float f) {
    union { float f; unsigned int u; } cv; cv.f = f;
    unsigned int u = cv.u;
    unsigned int r = (u + 0x7FFFu + ((u >> 16) & 1u)) >> 16;   // round-to-nearest-even
    return (unsigned short)r;
}

// Raw 32-bit LDS byte offset of a generic pointer known to point into LDS.
__device__ __forceinline__ unsigned lds_off(const void* p) {
    return (unsigned)(unsigned long long)(__attribute__((address_space(3))) const void*)p;
}

// Async global->LDS 16B copy (per-lane), tracked by ASYNCcnt.
__device__ __forceinline__ void async_copy16(unsigned lds_addr, const void* gaddr) {
    asm volatile("global_load_async_to_lds_b128 %0, %1, off"
                 :
                 : "v"(lds_addr), "v"(gaddr)
                 : "memory");
}

__device__ __forceinline__ void wait_async0() {
    asm volatile("s_wait_asynccnt 0x0" ::: "memory");
}

// ---------------------------------------------------------------------------
// Weight convert + transpose: W[K][N] fp32  ->  Wt[N][K] bf16
// ---------------------------------------------------------------------------
__global__ void convT_kernel(const float* __restrict__ w, unsigned short* __restrict__ wt,
                             int K, int N) {
    int e = blockIdx.x * 256 + threadIdx.x;
    if (e >= K * N) return;
    int k = e / N, n = e % N;
    wt[(size_t)n * K + k] = f2bf(w[e]);
}

// ---------------------------------------------------------------------------
// LayerNorm (row length 768) fused with fp32 -> bf16 cast. One wave per row.
// ---------------------------------------------------------------------------
__global__ __launch_bounds__(256) void ln_bf16_kernel(
    const float* __restrict__ x, const float* __restrict__ g, const float* __restrict__ b,
    unsigned short* __restrict__ out, int rows) {
    int wave = threadIdx.x >> 5, lane = threadIdx.x & 31;
    int row = blockIdx.x * 8 + wave;
    if (row >= rows) return;
    const float* xr = x + (size_t)row * DIM;
    float v[24];
    float s = 0.f;
    #pragma unroll
    for (int j = 0; j < 24; j++) { v[j] = xr[j * 32 + lane]; s += v[j]; }
    #pragma unroll
    for (int m = 16; m >= 1; m >>= 1) s += __shfl_xor(s, m, 32);
    float mu = s * (1.0f / DIM);
    float s2 = 0.f;
    #pragma unroll
    for (int j = 0; j < 24; j++) { float d = v[j] - mu; s2 += d * d; }
    #pragma unroll
    for (int m = 16; m >= 1; m >>= 1) s2 += __shfl_xor(s2, m, 32);
    float rstd = rsqrtf(s2 * (1.0f / DIM) + 1e-5f);
    unsigned short* orow = out + (size_t)row * DIM;
    #pragma unroll
    for (int j = 0; j < 24; j++) {
        int c = j * 32 + lane;
        orow[c] = f2bf((v[j] - mu) * rstd * g[c] + b[c]);
    }
}

// ---------------------------------------------------------------------------
// BF16 WMMA GEMM: out = A[M][K](bf16) * Wt[N][K](bf16, pre-transposed) + bias
// Block: 256 threads (8 waves), tile 128(M) x 128(N); wave tile 32x64
// (2x4 WMMA per K-step). Double-buffered LDS filled by async global->LDS.
// MODE 0: bf16 out (bias)        MODE 1: fp32 out (bias + residual)
// MODE 2: bf16 out (bias + exact GELU)
// ---------------------------------------------------------------------------
template <int MODE>
__global__ __launch_bounds__(256) void gemm_bf16_kernel(
    const unsigned short* __restrict__ A, const unsigned short* __restrict__ Wt,
    const float* __restrict__ bias, const float* __restrict__ resid,
    float* __restrict__ outf, unsigned short* __restrict__ outb,
    int M, int N, int K) {
    __shared__ __align__(16) unsigned short a_lds[2][128 * 32];   // 2 x 8 KB
    __shared__ __align__(16) unsigned short b_lds[2][128 * 32];   // 2 x 8 KB

    int tid = threadIdx.x, lane = tid & 31, wid = tid >> 5;
    int h = lane >> 4, lm = lane & 15;
    int m0 = blockIdx.y * 128, n0 = blockIdx.x * 128;
    int wm = wid >> 1, wn = wid & 1;          // 4x2 wave grid -> 32(M) x 64(N) per wave

    v8f acc[2][4] = {};

    // issue async loads of one K-slab (A: 128x32, B: 128x32) into buffer `buf`
    auto prefetch = [&](int buf, int k0) {
        #pragma unroll
        for (int i = 0; i < 2; i++) {
            int c = tid + 256 * i;
            int r = c >> 2, off = (c & 3) * 8;
            async_copy16(lds_off(&a_lds[buf][r * 32 + off]),
                         &A[(size_t)(m0 + r) * K + k0 + off]);
        }
        #pragma unroll
        for (int i = 0; i < 2; i++) {
            int c = tid + 256 * i;
            int n = c >> 2, off = (c & 3) * 8;
            async_copy16(lds_off(&b_lds[buf][n * 32 + off]),
                         &Wt[(size_t)(n0 + n) * K + k0 + off]);
        }
    };

    prefetch(0, 0);
    wait_async0();
    __syncthreads();

    int buf = 0;
    for (int k0 = 0; k0 < K; k0 += 32, buf ^= 1) {
        if (k0 + 32 < K) prefetch(buf ^ 1, k0 + 32);   // overlap next slab with compute

        const unsigned short* al = a_lds[buf];
        const unsigned short* bl = b_lds[buf];

        BF16Frag af[2];
        #pragma unroll
        for (int mi = 0; mi < 2; mi++) {
            int m = wm * 32 + mi * 16 + lm;
            af[mi].u[0] = *(const uint4*)&al[m * 32 + 8 * h];
            af[mi].u[1] = *(const uint4*)&al[m * 32 + 16 + 8 * h];
        }
        #pragma unroll
        for (int ni = 0; ni < 4; ni++) {
            BF16Frag bf;
            int n = wn * 64 + ni * 16 + lm;
            bf.u[0] = *(const uint4*)&bl[n * 32 + 8 * h];
            bf.u[1] = *(const uint4*)&bl[n * 32 + 16 + 8 * h];
            #pragma unroll
            for (int mi = 0; mi < 2; mi++)
                acc[mi][ni] = __builtin_amdgcn_wmma_f32_16x16x32_bf16(
                    false, af[mi].v, false, bf.v, (short)0, acc[mi][ni], false, false);
        }

        wait_async0();       // next slab landed in LDS (this wave's portion)
        __syncthreads();     // all waves' slabs landed, all done reading `buf`
    }

    // epilogue: C/D layout -> element i is row (i + 8h), column lm of the tile
    #pragma unroll
    for (int mi = 0; mi < 2; mi++) {
        #pragma unroll
        for (int ni = 0; ni < 4; ni++) {
            int gn = n0 + wn * 64 + ni * 16 + lm;
            float bia = bias[gn];
            #pragma unroll
            for (int i = 0; i < 8; i++) {
                int gm = m0 + wm * 32 + mi * 16 + 8 * h + i;
                float v = acc[mi][ni][i] + bia;
                size_t o = (size_t)gm * N + gn;
                if (MODE == 0) {
                    outb[o] = f2bf(v);
                } else if (MODE == 1) {
                    outf[o] = resid[o] + v;
                } else {
                    float gg = 0.5f * v * (1.0f + erff(v * 0.70710678118654752f));
                    outb[o] = f2bf(gg);
                }
            }
        }
    }
}

// ---------------------------------------------------------------------------
// Flash attention, bf16 WMMA. 128 threads = 4 waves; each wave owns 16 q rows
// (block covers 64 q rows of one (batch, head)). Online softmax over key
// blocks of 32. qkv layout: [ROWS][2304] with q|k|v each [head][64].
// ---------------------------------------------------------------------------
__global__ __launch_bounds__(128) void attn_kernel(
    const unsigned short* __restrict__ qkv, unsigned short* __restrict__ o) {
    __shared__ __align__(16) unsigned short k_lds[32 * 64];   // K block, row-major
    __shared__ __align__(16) unsigned short vt_lds[64 * 32];  // V block, transposed [d][key]
    __shared__ __align__(16) unsigned short p_lds[4 * 16 * 32];

    int tid = threadIdx.x, lane = tid & 31, wave = tid >> 5;
    int h = lane >> 4, lm = lane & 15;
    int bh = blockIdx.y;
    int b = bh / HEADS, head = bh % HEADS;
    int qblk = blockIdx.x;
    size_t rowbase = (size_t)b * SEQ;
    int qcol = head * HEAD_DIM;
    int kcol = DIM + head * HEAD_DIM;
    int vcol = 2 * DIM + head * HEAD_DIM;

    // Q fragments (16x64 per wave -> two 16x32 A fragments), resident all kernel
    int qm = qblk * 64 + wave * 16 + lm;
    const unsigned short* qrow = qkv + (rowbase + qm) * 2304 + qcol;
    BF16Frag qf[2];
    #pragma unroll
    for (int kk = 0; kk < 2; kk++) {
        qf[kk].u[0] = *(const uint4*)&qrow[32 * kk + 8 * h];
        qf[kk].u[1] = *(const uint4*)&qrow[32 * kk + 16 + 8 * h];
    }

    float mrow[8], lrow[8];
    v8f accd[4] = {};
    #pragma unroll
    for (int i = 0; i < 8; i++) { mrow[i] = -1e30f; lrow[i] = 0.f; }
    const float scale = 0.125f;            // 1/sqrt(64)
    const float LOG2E = 1.44269504088896f;

    for (int kb = 0; kb < SEQ / 32; kb++) {
        int j0 = kb * 32;

        // prefetch next key block (K and V rows) into cache
        if (kb + 1 < SEQ / 32) {
            int pr = tid >> 2, pc = (tid & 3) * 16;
            const unsigned short* nrow = &qkv[(rowbase + j0 + 32 + pr) * 2304];
            __builtin_prefetch(nrow + kcol + pc, 0, 0);
            __builtin_prefetch(nrow + vcol + pc, 0, 0);
        }

        __syncthreads();   // protect LDS K/V reuse across iterations
        // K block 32x64, row-major
        #pragma unroll
        for (int i = 0; i < 2; i++) {
            int c = tid + 128 * i;
            int kr = c >> 3, off = (c & 7) * 8;
            *(uint4*)&k_lds[kr * 64 + off] =
                *(const uint4*)&qkv[(rowbase + j0 + kr) * 2304 + kcol + off];
        }
        // V block, transposed into LDS [d][key]
        {
            int vr = tid >> 2;
            int doff = (tid & 3) * 16;
            const unsigned short* vsrc = &qkv[(rowbase + j0 + vr) * 2304 + vcol + doff];
            union { uint4 u[2]; unsigned short s[16]; } tv;
            tv.u[0] = *(const uint4*)vsrc;
            tv.u[1] = *(const uint4*)(vsrc + 8);
            #pragma unroll
            for (int e = 0; e < 16; e++) vt_lds[(doff + e) * 32 + vr] = tv.s[e];
        }
        __syncthreads();

        // S = Q * K^T  (two 16x16 tiles across 32 keys)
        v8f s[2];
        #pragma unroll
        for (int jt = 0; jt < 2; jt++) {
            v8f c = {};
            #pragma unroll
            for (int kk = 0; kk < 2; kk++) {
                BF16Frag kf;
                int jl = jt * 16 + lm;
                kf.u[0] = *(const uint4*)&k_lds[jl * 64 + 32 * kk + 8 * h];
                kf.u[1] = *(const uint4*)&k_lds[jl * 64 + 32 * kk + 16 + 8 * h];
                c = __builtin_amdgcn_wmma_f32_16x16x32_bf16(
                    false, qf[kk].v, false, kf.v, (short)0, c, false, false);
            }
            s[jt] = c;
        }

        // online softmax; row r = i + 8h lives in one 16-lane half-group
        float mloc[8];
        #pragma unroll
        for (int i = 0; i < 8; i++) {
            s[0][i] *= scale; s[1][i] *= scale;
            mloc[i] = fmaxf(s[0][i], s[1][i]);
        }
        #pragma unroll
        for (int m = 1; m < 16; m <<= 1)
            #pragma unroll
            for (int i = 0; i < 8; i++) mloc[i] = fmaxf(mloc[i], __shfl_xor(mloc[i], m, 32));
        float psum[8];
        #pragma unroll
        for (int i = 0; i < 8; i++) {
            float mn = fmaxf(mrow[i], mloc[i]);
            float alpha = exp2f((mrow[i] - mn) * LOG2E);
            mrow[i] = mn;
            s[0][i] = exp2f((s[0][i] - mn) * LOG2E);
            s[1][i] = exp2f((s[1][i] - mn) * LOG2E);
            psum[i] = s[0][i] + s[1][i];
            lrow[i] *= alpha;
            #pragma unroll
            for (int dt = 0; dt < 4; dt++) accd[dt][i] *= alpha;
        }
        #pragma unroll
        for (int m = 1; m < 16; m <<= 1)
            #pragma unroll
            for (int i = 0; i < 8; i++) psum[i] += __shfl_xor(psum[i], m, 32);
        #pragma unroll
        for (int i = 0; i < 8; i++) lrow[i] += psum[i];

        // C-layout -> A-fragment layout via LDS (per-wave region; same-wave LDS
        // ops are processed in order, so the RAW is safe without a barrier)
        #pragma unroll
        for (int jt = 0; jt < 2; jt++)
            #pragma unroll
            for (int i = 0; i < 8; i++)
                p_lds[wave * 512 + (8 * h + i) * 32 + jt * 16 + lm] = f2bf(s[jt][i]);
        BF16Frag pf;
        pf.u[0] = *(const uint4*)&p_lds[wave * 512 + lm * 32 + 8 * h];
        pf.u[1] = *(const uint4*)&p_lds[wave * 512 + lm * 32 + 16 + 8 * h];

        // acc += P * V  (B fragments from transposed V: contiguous reads)
        #pragma unroll
        for (int dt = 0; dt < 4; dt++) {
            BF16Frag vf;
            int dl = dt * 16 + lm;
            vf.u[0] = *(const uint4*)&vt_lds[dl * 32 + 8 * h];
            vf.u[1] = *(const uint4*)&vt_lds[dl * 32 + 16 + 8 * h];
            accd[dt] = __builtin_amdgcn_wmma_f32_16x16x32_bf16(
                false, pf.v, false, vf.v, (short)0, accd[dt], false, false);
        }
    }

    // normalize and store bf16 output [ROWS][768]
    #pragma unroll
    for (int dt = 0; dt < 4; dt++) {
        int gc = head * HEAD_DIM + dt * 16 + lm;
        #pragma unroll
        for (int i = 0; i < 8; i++) {
            int gr = qblk * 64 + wave * 16 + 8 * h + i;
            o[(rowbase + gr) * DIM + gc] = f2bf(accd[dt][i] / lrow[i]);
        }
    }
}

// ---------------------------------------------------------------------------
extern "C" void kernel_launch(void* const* d_in, const int* in_sizes, int n_in,
                              void* d_out, int out_size, void* d_ws, size_t ws_size,
                              hipStream_t stream) {
    (void)in_sizes; (void)n_in; (void)out_size; (void)ws_size;
    const float* x      = (const float*)d_in[0];
    const float* ln1_g  = (const float*)d_in[1];
    const float* ln1_b  = (const float*)d_in[2];
    const float* qkv_w  = (const float*)d_in[3];
    const float* qkv_b  = (const float*)d_in[4];
    const float* proj_w = (const float*)d_in[5];
    const float* proj_b = (const float*)d_in[6];
    const float* ln2_g  = (const float*)d_in[7];
    const float* ln2_b  = (const float*)d_in[8];
    const float* fc1_w  = (const float*)d_in[9];
    const float* fc1_b  = (const float*)d_in[10];
    const float* fc2_w  = (const float*)d_in[11];
    const float* fc2_b  = (const float*)d_in[12];
    float* out = (float*)d_out;

    // workspace layout (bytes)
    char* ws = (char*)d_ws;
    size_t off = 0;
    unsigned short* wt_qkv  = (unsigned short*)(ws + off); off += (size_t)2304 * 768 * 2;  // 3.5 MB
    unsigned short* wt_proj = (unsigned short*)(ws + off); off += (size_t)768 * 768 * 2;   // 1.2 MB
    unsigned short* wt_fc1  = (unsigned short*)(ws + off); off += (size_t)3072 * 768 * 2;  // 4.7 MB
    unsigned short* wt_fc2  = (unsigned short*)(ws + off); off += (size_t)768 * 3072 * 2;  // 4.7 MB
    unsigned short* h_bf    = (unsigned short*)(ws + off); off += (size_t)ROWS * DIM * 2;  // 12.6 MB
    unsigned short* qkv_bf  = (unsigned short*)(ws + off); off += (size_t)ROWS * 2304 * 2; // 37.7 MB (reused for gelu out)
    unsigned short* o_bf    = (unsigned short*)(ws + off); off += (size_t)ROWS * DIM * 2;  // 12.6 MB
    float*          x1      = (float*)(ws + off);          off += (size_t)ROWS * DIM * 4;  // 25.2 MB

    // one-time weight bf16 transposes
    convT_kernel<<<(768 * 2304 + 255) / 256, 256, 0, stream>>>(qkv_w, wt_qkv, 768, 2304);
    convT_kernel<<<(768 * 768 + 255) / 256, 256, 0, stream>>>(proj_w, wt_proj, 768, 768);
    convT_kernel<<<(768 * 3072 + 255) / 256, 256, 0, stream>>>(fc1_w, wt_fc1, 768, 3072);
    convT_kernel<<<(3072 * 768 + 255) / 256, 256, 0, stream>>>(fc2_w, wt_fc2, 3072, 768);

    // LN1 -> bf16
    ln_bf16_kernel<<<ROWS / 8, 256, 0, stream>>>(x, ln1_g, ln1_b, h_bf, ROWS);
    // QKV gemm: [8192x768] x [768x2304]
    gemm_bf16_kernel<0><<<dim3(2304 / 128, ROWS / 128), 256, 0, stream>>>(
        h_bf, wt_qkv, qkv_b, nullptr, nullptr, qkv_bf, ROWS, 2304, 768);
    // attention
    attn_kernel<<<dim3(SEQ / 64, BATCH * HEADS), 128, 0, stream>>>(qkv_bf, o_bf);
    // proj gemm + residual: x1 = x + o @ proj_w + b
    gemm_bf16_kernel<1><<<dim3(768 / 128, ROWS / 128), 256, 0, stream>>>(
        o_bf, wt_proj, proj_b, x, x1, nullptr, ROWS, 768, 768);
    // LN2 -> bf16
    ln_bf16_kernel<<<ROWS / 8, 256, 0, stream>>>(x1, ln2_g, ln2_b, h_bf, ROWS);
    // FC1 + exact GELU -> bf16 (reuse qkv buffer)
    gemm_bf16_kernel<2><<<dim3(3072 / 128, ROWS / 128), 256, 0, stream>>>(
        h_bf, wt_fc1, fc1_b, nullptr, nullptr, qkv_bf, ROWS, 3072, 768);
    // FC2 + residual -> fp32 output
    gemm_bf16_kernel<1><<<dim3(768 / 128, ROWS / 128), 256, 0, stream>>>(
        qkv_bf, wt_fc2, fc2_b, x1, out, nullptr, ROWS, 768, 3072);
}
